// GraphSAGE_17506286698960
// MI455X (gfx1250) — matrix operand order repaired and checked by reading, not compiled, not run on hardware
//
#include <hip/hip_runtime.h>
#include <hip/hip_bf16.h>

typedef __attribute__((ext_vector_type(16))) __bf16 v16bf;
typedef __attribute__((ext_vector_type(8)))  float  v8f;

// ---------------------------------------------------------------------------
// Utility kernels
// ---------------------------------------------------------------------------
__global__ void zero_kernel(float* __restrict__ p, long n) {
  long i = (long)blockIdx.x * blockDim.x + threadIdx.x;
  long stride = (long)gridDim.x * blockDim.x;
  for (; i < n; i += stride) p[i] = 0.0f;
}

// Pack fp32 weight [128][Hout] (k-major) into transposed bf16 [Hout][128]
// so WMMA B-fragments are contiguous 16-bf16 (32B) runs per lane.
__global__ void pack_weight_kernel(const float* __restrict__ W,
                                   __bf16* __restrict__ WT, int Hout) {
  int idx = blockIdx.x * blockDim.x + threadIdx.x;
  if (idx < 128 * Hout) {
    int k = idx / Hout;
    int n = idx % Hout;
    WT[n * 128 + k] = (__bf16)W[idx];
  }
}

__global__ void degree_kernel(const int* __restrict__ dst,
                              float* __restrict__ deg, int E) {
  int e = blockIdx.x * blockDim.x + threadIdx.x;
  if (e < E) atomicAdd(&deg[dst[e]], 1.0f);
}

__global__ void recip_kernel(const float* __restrict__ deg,
                             float* __restrict__ dinv, int n) {
  int i = blockIdx.x * blockDim.x + threadIdx.x;
  if (i < n) dinv[i] = 1.0f / fmaxf(deg[i], 1.0f);
}

// One wave (32 lanes) per edge; lane handles 4 contiguous floats (float4
// gather from L2-resident feature table, fp32 atomics into L2-resident agg).
__global__ void aggregate_kernel(const float* __restrict__ feat,
                                 const int* __restrict__ src,
                                 const int* __restrict__ dst,
                                 float* __restrict__ agg, int E) {
  int edge = blockIdx.x * (blockDim.x >> 5) + (threadIdx.x >> 5);
  if (edge >= E) return;
  int lane = threadIdx.x & 31;
  int s = src[edge];
  int d = dst[edge];
  const float4 v = *(const float4*)(feat + (size_t)s * 128 + lane * 4);
  float* o = agg + (size_t)d * 128 + lane * 4;
  atomicAdd(o + 0, v.x);
  atomicAdd(o + 1, v.y);
  atomicAdd(o + 2, v.z);
  atomicAdd(o + 3, v.w);
}

// ---------------------------------------------------------------------------
// Fused SAGE layer:  out[16 nodes x HOUT] = X@Wself + (AGG*dinv)@Wneigh + b
// One block = 16 nodes; wave w computes output columns [w*16, w*16+16).
// K = 128, consumed as 4 x wmma_f32_16x16x32_bf16 per weight matrix.
// ---------------------------------------------------------------------------
template <int HOUT, bool RELU>
__global__ __launch_bounds__((HOUT / 16) * 32)
void sage_layer_kernel(const float*  __restrict__ X,     // [N][128]
                       const float*  __restrict__ AGG,   // [N][128] (sum)
                       const float*  __restrict__ DINV,  // [N]
                       const __bf16* __restrict__ WTS,   // [HOUT][128] bf16^T
                       const __bf16* __restrict__ WTN,   // [HOUT][128] bf16^T
                       const float*  __restrict__ BIAS,  // [HOUT]
                       float*        __restrict__ OUT,   // [N][HOUT]
                       int N) {
  constexpr int KIN = 128;
  constexpr int NTH = (HOUT / 16) * 32;

  __shared__ __bf16 a_self [16][KIN];   // 4 KB
  __shared__ __bf16 a_neigh[16][KIN];   // 4 KB

  const int tid   = threadIdx.x;
  const int node0 = blockIdx.x * 16;

  // Stage 16-node activation tile as bf16; apply mean scaling to neighbor agg.
  for (int idx = tid; idx < 16 * KIN; idx += NTH) {
    int r = idx / KIN;
    int k = idx % KIN;
    int node = node0 + r;
    float xv = 0.0f, av = 0.0f;
    if (node < N) {
      xv = X[(size_t)node * KIN + k];
      av = AGG[(size_t)node * KIN + k] * DINV[node];
    }
    a_self [r][k] = (__bf16)xv;
    a_neigh[r][k] = (__bf16)av;
  }
  __syncthreads();

  const int wave = tid >> 5;
  const int lane = tid & 31;
  const int half = lane >> 4;   // 0: lanes 0-15, 1: lanes 16-31
  const int rm   = lane & 15;
  const int col  = wave * 16 + rm;

  v8f c = {};

  #pragma unroll
  for (int kt = 0; kt < KIN / 32; ++kt) {
    const int kbase = kt * 32;

    // A fragment (16x32 bf16): lane half h holds K = kbase + h*16 + {0..7}
    // in elems 0..7 and K = kbase + h*16 + 8 + {0..7} in elems 8..15.
    v16bf as, an;
    const __bf16* arow_s = &a_self [rm][0];
    const __bf16* arow_n = &a_neigh[rm][0];
    const int ka = kbase + half * 16;
    #pragma unroll
    for (int e = 0; e < 8; ++e) {
      as[e]     = arow_s[ka + e];
      as[e + 8] = arow_s[ka + 8 + e];
      an[e]     = arow_n[ka + e];
      an[e + 8] = arow_n[ka + 8 + e];
    }

    // B fragment (32x16 bf16): lane (n = rm) holds K = kbase + half*16 + e,
    // contiguous in the transposed packed weight rows.
    v16bf bs, bn;
    const __bf16* brow_s = WTS + (size_t)col * KIN;
    const __bf16* brow_n = WTN + (size_t)col * KIN;
    const int kb = kbase + half * 16;
    #pragma unroll
    for (int e = 0; e < 16; ++e) {
      bs[e] = brow_s[kb + e];
      bn[e] = brow_n[kb + e];
    }

    // D = A*B + C with fp32 accumulation; fuse self + neighbor chains.
    c = __builtin_amdgcn_wmma_f32_16x16x32_bf16(false, as, false, bs,
                                                (short)0, c, false, false);
    c = __builtin_amdgcn_wmma_f32_16x16x32_bf16(false, an, false, bn,
                                                (short)0, c, false, false);
  }

  // Epilogue: bias (+ReLU). C/D layout: VGPR i -> row (half*8 + i), col = col.
  const float bb = BIAS[col];
  #pragma unroll
  for (int i = 0; i < 8; ++i) {
    int row  = half * 8 + i;
    int node = node0 + row;
    if (node < N) {
      float v = c[i] + bb;
      if (RELU) v = fmaxf(v, 0.0f);
      OUT[(size_t)node * HOUT + col] = v;
    }
  }
}

// ---------------------------------------------------------------------------
// Launch: deg -> dinv -> agg(x) -> layer1(WMMA,relu) -> agg(h1) -> layer3
// Layer 2 is dead code in the reference (h2 = relu(h1) = h1) and is skipped.
// ---------------------------------------------------------------------------
extern "C" void kernel_launch(void* const* d_in, const int* in_sizes, int n_in,
                              void* d_out, int out_size, void* d_ws, size_t ws_size,
                              hipStream_t stream) {
  const float* x    = (const float*)d_in[0];
  const int*   esrc = (const int*)  d_in[1];
  const int*   edst = (const int*)  d_in[2];
  const float* ws1  = (const float*)d_in[3];
  const float* wn1  = (const float*)d_in[4];
  const float* b1   = (const float*)d_in[5];
  // d_in[6..8] = layer-2 params: dead in the reference, intentionally unused.
  const float* ws3  = (const float*)d_in[9];
  const float* wn3  = (const float*)d_in[10];
  const float* b3   = (const float*)d_in[11];
  float* out = (float*)d_out;

  const int N = in_sizes[0] / 128;
  const int E = in_sizes[1];

  // Workspace layout (floats): deg | dinv | agg[128N] | h1[128N] | bf16 weights
  float* ws   = (float*)d_ws;
  float* deg  = ws;
  float* dinv = ws + (size_t)N;
  float* agg  = ws + (size_t)2 * N;
  float* h1   = ws + (size_t)2 * N + (size_t)128 * N;
  __bf16* wpack = (__bf16*)(ws + (size_t)2 * N + (size_t)256 * N);
  __bf16* wt_s1 = wpack;
  __bf16* wt_n1 = wt_s1 + 128 * 128;
  __bf16* wt_s3 = wt_n1 + 128 * 128;
  __bf16* wt_n3 = wt_s3 + 64 * 128;

  // Zero deg + dinv + agg in one sweep ([0, 130N)); dinv overwritten below.
  zero_kernel<<<2048, 256, 0, stream>>>(ws, (long)130 * N);

  pack_weight_kernel<<<(128 * 128 + 255) / 256, 256, 0, stream>>>(ws1, wt_s1, 128);
  pack_weight_kernel<<<(128 * 128 + 255) / 256, 256, 0, stream>>>(wn1, wt_n1, 128);
  pack_weight_kernel<<<(128 * 64  + 255) / 256, 256, 0, stream>>>(ws3, wt_s3, 64);
  pack_weight_kernel<<<(128 * 64  + 255) / 256, 256, 0, stream>>>(wn3, wt_n3, 64);

  degree_kernel<<<(E + 255) / 256, 256, 0, stream>>>(edst, deg, E);
  recip_kernel <<<(N + 255) / 256, 256, 0, stream>>>(deg, dinv, N);

  // Layer 1
  aggregate_kernel<<<(E + 7) / 8, 256, 0, stream>>>(x, esrc, edst, agg, E);
  const int gblocks = (N + 15) / 16;
  sage_layer_kernel<128, true><<<gblocks, 256, 0, stream>>>(
      x, agg, dinv, wt_s1, wt_n1, b1, h1, N);

  // Layer 3 (layer 2 skipped: h2 == h1 per reference bug)
  zero_kernel<<<2048, 256, 0, stream>>>(agg, (long)128 * N);
  aggregate_kernel<<<(E + 7) / 8, 256, 0, stream>>>(h1, esrc, edst, agg, E);
  sage_layer_kernel<64, false><<<gblocks, 128, 0, stream>>>(
      h1, agg, dinv, wt_s3, wt_n3, b3, out, N);
}
